// LLaMA_31988916420692
// MI455X (gfx1250) — matrix operand order repaired and checked by reading, not compiled, not run on hardware
//
#include <hip/hip_runtime.h>
#include <cmath>

typedef __bf16 bf16;
typedef __attribute__((ext_vector_type(16))) __bf16 bf16x16;
typedef __attribute__((ext_vector_type(8)))  float  f32x8;

#if defined(__has_builtin)
#  if __has_builtin(__builtin_amdgcn_sched_barrier)
#    define SCHED_FENCE() __builtin_amdgcn_sched_barrier(0)
#  endif
#endif
#ifndef SCHED_FENCE
#  define SCHED_FENCE()
#endif

constexpr int kD   = 2048;
constexpr int kNH  = 32;
constexpr int kNKV = 8;
constexpr int kHD  = 64;
constexpr int kF   = 5632;
constexpr int kV   = 32000;
constexpr int kL   = 2;
constexpr int kB   = 2;
constexpr int kS   = 1024;
constexpr int kNT  = kB * kS;                    // 2048 tokens
constexpr int kQKV = (kNH + 2 * kNKV) * kHD;     // 3072
constexpr int kG   = kNH / kNKV;                 // 4

// ---------------------------------------------------------------------------
// Generic batched WMMA GEMM:  C[M,N] (+)= A[M,K] * W[N,K]^T   (bf16 in, f32 out)
// A, W row-major, leading dim K; C leading dim N.
// K/32 must be EVEN (all shapes here: nk in {2,32,64,176}).
// Two-deep software pipeline: global loads for block k+2 are issued while
// block k computes; double-buffered LDS; one barrier per k-step.
// Out-of-range tile rows are CLAMPED (not zeroed): they only affect output
// rows/cols that the guarded epilogue never stores.
// ---------------------------------------------------------------------------
#define BM 128
#define BN 128
#define BK 32
#define LP 40   // LDS row pitch in bf16 (40*2=80B -> conflict-free b128 reads)

__global__ __launch_bounds__(256)
void gemm_bf16_kernel(float* __restrict__ C, const bf16* __restrict__ A,
                      const bf16* __restrict__ W,
                      int M, int N, int K,
                      long sC, long sA, long sW, int accum, int causal)
{
    int bn = blockIdx.x * BN;
    int bm = blockIdx.y * BM;
    if (causal && bn >= bm + BM) return;   // entire tile above the diagonal

    long batch = blockIdx.z;
    C += batch * sC;  A += batch * sA;  W += batch * sW;

    __shared__ __align__(16) bf16 sAt[2][BM * LP];
    __shared__ __align__(16) bf16 sWt[2][BN * LP];

    const int tid  = threadIdx.x;
    const int lane = tid & 31;
    const int wave = tid >> 5;
    const int waveM = wave >> 2;      // 0..1 -> 64 rows each
    const int waveN = wave & 3;       // 0..3 -> 32 cols each
    const int l15 = lane & 15;
    const int lh  = lane >> 4;

    // --- tile staging: each thread owns 2 rows (r, r+64) x one 16B segment ---
    const int seg  = (tid & 3) * 8;       // bf16 offset within 32-elem row
    const int row0 = tid >> 2;            // 0..63
    const int rowA0 = (bm + row0      < M) ? bm + row0      : M - 1;
    const int rowA1 = (bm + row0 + 64 < M) ? bm + row0 + 64 : M - 1;
    const int rowW0 = (bn + row0      < N) ? bn + row0      : N - 1;
    const int rowW1 = (bn + row0 + 64 < N) ? bn + row0 + 64 : N - 1;
    const bf16* pA0 = A + (long)rowA0 * K + seg;
    const bf16* pA1 = A + (long)rowA1 * K + seg;
    const bf16* pW0 = W + (long)rowW0 * K + seg;
    const bf16* pW1 = W + (long)rowW1 * K + seg;
    const int stO0 = row0 * LP + seg;
    const int stO1 = (row0 + 64) * LP + seg;

    // Two register staging sets -> 2-deep global prefetch queue.
    uint4 q0a0, q0a1, q0w0, q0w1;   // set 0
    uint4 q1a0, q1a1, q1w0, q1w1;   // set 1
    auto fetch0 = [&]() {
        q0a0 = *(const uint4*)pA0;  q0a1 = *(const uint4*)pA1;
        q0w0 = *(const uint4*)pW0;  q0w1 = *(const uint4*)pW1;
        pA0 += BK;  pA1 += BK;  pW0 += BK;  pW1 += BK;
    };
    auto fetch1 = [&]() {
        q1a0 = *(const uint4*)pA0;  q1a1 = *(const uint4*)pA1;
        q1w0 = *(const uint4*)pW0;  q1w1 = *(const uint4*)pW1;
        pA0 += BK;  pA1 += BK;  pW0 += BK;  pW1 += BK;
    };
    auto stage0 = [&](int buf) {
        *(uint4*)(sAt[buf] + stO0) = q0a0;
        *(uint4*)(sAt[buf] + stO1) = q0a1;
        *(uint4*)(sWt[buf] + stO0) = q0w0;
        *(uint4*)(sWt[buf] + stO1) = q0w1;
    };
    auto stage1 = [&](int buf) {
        *(uint4*)(sAt[buf] + stO0) = q1a0;
        *(uint4*)(sAt[buf] + stO1) = q1a1;
        *(uint4*)(sWt[buf] + stO0) = q1w0;
        *(uint4*)(sWt[buf] + stO1) = q1w1;
    };

    f32x8 acc[4][2];
    f32x8 zero = {0.f,0.f,0.f,0.f,0.f,0.f,0.f,0.f};
#pragma unroll
    for (int mi = 0; mi < 4; ++mi)
#pragma unroll
        for (int ni = 0; ni < 2; ++ni) acc[mi][ni] = zero;

    auto compute = [&](int buf) {
        union FR { bf16x16 v; uint4 u[2]; } wf[2], af;
        // B operand: lane holds output-col (l15), contiguous K chunk lh*16..+15
#pragma unroll
        for (int ni = 0; ni < 2; ++ni) {
            const bf16* p = sWt[buf] + (waveN * 32 + ni * 16 + l15) * LP + lh * 16;
            wf[ni].u[0] = *(const uint4*)(p);
            wf[ni].u[1] = *(const uint4*)(p + 8);
        }
        // A operand: lane holds row (l15), contiguous K chunk lh*16..+15
#pragma unroll
        for (int mi = 0; mi < 4; ++mi) {
            const bf16* p = sAt[buf] + (waveM * 64 + mi * 16 + l15) * LP + lh * 16;
            af.u[0] = *(const uint4*)(p);
            af.u[1] = *(const uint4*)(p + 8);
#pragma unroll
            for (int ni = 0; ni < 2; ++ni) {
                acc[mi][ni] = __builtin_amdgcn_wmma_f32_16x16x32_bf16(
                    false, af.v, false, wf[ni].v, (short)0, acc[mi][ni],
                    false, false);
            }
        }
    };

    // --- 2-deep pipeline (nk even): fetch(k+2) || wmma(k) ---
    const int nk = K / BK;
    fetch0();                 // block 0 -> set0
    fetch1();                 // block 1 -> set1
    stage0(0);
    __syncthreads();
    int buf = 0;
    for (int kb = 1; kb < nk - 1; kb += 2) {
        fetch0();             // block kb+1 (in flight through this compute)
        SCHED_FENCE();
        compute(buf);         // block kb-1
        stage1(buf ^ 1);      // block kb   (loads issued one iter ago)
        __syncthreads();
        buf ^= 1;
        fetch1();             // block kb+2
        SCHED_FENCE();
        compute(buf);         // block kb
        stage0(buf ^ 1);      // block kb+1
        __syncthreads();
        buf ^= 1;
    }
    compute(buf);             // block nk-2
    stage1(buf ^ 1);          // block nk-1
    __syncthreads();
    compute(buf ^ 1);         // block nk-1

    // Epilogue: C/D layout -> m = 8*(lane>>4)+i, n = lane&15
#pragma unroll
    for (int mi = 0; mi < 4; ++mi) {
#pragma unroll
        for (int ni = 0; ni < 2; ++ni) {
            int col = bn + waveN * 32 + ni * 16 + l15;
            if (col < N) {
#pragma unroll
                for (int i = 0; i < 8; ++i) {
                    long row = (long)bm + waveM * 64 + mi * 16 + lh * 8 + i;
                    if (row < M) {
                        long off = row * (long)N + col;
                        float v = acc[mi][ni][i];
                        if (accum) v += C[off];
                        C[off] = v;
                    }
                }
            }
        }
    }
}

// ---------------------------------------------------------------------------
// Elementwise / helper kernels
// ---------------------------------------------------------------------------
__global__ void embed_kernel(const int* __restrict__ ids,
                             const float* __restrict__ emb,
                             float* __restrict__ x)
{
    int n = blockIdx.x;
    long src = (long)ids[n] * kD;
    long dst = (long)n * kD;
    for (int i = threadIdx.x * 4; i < kD; i += blockDim.x * 4)
        *(float4*)(x + dst + i) = *(const float4*)(emb + src + i);
}

__global__ void cvt_f32_bf16_kernel(const float* __restrict__ in,
                                    bf16* __restrict__ out, long n)
{
    long i = ((long)blockIdx.x * blockDim.x + threadIdx.x) * 4;
    if (i < n) {
        float4 v = *(const float4*)(in + i);
        out[i + 0] = (bf16)v.x;  out[i + 1] = (bf16)v.y;
        out[i + 2] = (bf16)v.z;  out[i + 3] = (bf16)v.w;
    }
}

__global__ __launch_bounds__(256)
void rmsnorm_kernel(const float* __restrict__ x, const float* __restrict__ w,
                    bf16* __restrict__ out)
{
    int row = blockIdx.x;
    const float* xr = x + (long)row * kD;
    float ss = 0.f;
    for (int i = threadIdx.x; i < kD; i += blockDim.x) {
        float v = xr[i];
        ss = fmaf(v, v, ss);
    }
    __shared__ float sh[9];
    int lane = threadIdx.x & 31, wave = threadIdx.x >> 5;
    for (int o = 16; o > 0; o >>= 1) ss += __shfl_down(ss, o, 32);
    if (lane == 0) sh[wave] = ss;
    __syncthreads();
    if (wave == 0) {
        float v = (lane < 8) ? sh[lane] : 0.f;
        for (int o = 4; o > 0; o >>= 1) v += __shfl_down(v, o, 32);
        if (lane == 0) sh[8] = v;
    }
    __syncthreads();
    float inv = rsqrtf(sh[8] / (float)kD + 1e-5f);
    for (int i = threadIdx.x; i < kD; i += blockDim.x)
        out[(long)row * kD + i] = (bf16)(xr[i] * inv * w[i]);
}

// RoPE on q/k, split into per-head bf16 buffers; GQA-replicate k; transpose v.
__global__ __launch_bounds__(256)
void rope_split_kernel(const float* __restrict__ qkv,
                       bf16* __restrict__ qb, bf16* __restrict__ kb,
                       bf16* __restrict__ vt)
{
    int n = blockIdx.x;
    int b = n / kS, spos = n % kS;
    const float* row = qkv + (long)n * kQKV;
    float pos = (float)spos;

    for (int p = threadIdx.x; p < (kNH + kNKV) * (kHD / 2); p += blockDim.x) {
        int head = p >> 5;     // HD/2 = 32 pairs per head
        int j = p & 31;
        int col = head * kHD + 2 * j;
        float x0 = row[col], x1 = row[col + 1];
        float inv = powf(10000.0f, -(2.0f * (float)j) / (float)kHD);
        float ang = pos * inv;
        float s, c;
        sincosf(ang, &s, &c);
        float o0 = x0 * c - x1 * s;
        float o1 = x1 * c + x0 * s;
        if (head < kNH) {
            long base = (((long)b * kNH + head) * kS + spos) * kHD + 2 * j;
            qb[base] = (bf16)o0;  qb[base + 1] = (bf16)o1;
        } else {
            int kh = head - kNH;
            for (int r = 0; r < kG; ++r) {
                int h = kh * kG + r;
                long base = (((long)b * kNH + h) * kS + spos) * kHD + 2 * j;
                kb[base] = (bf16)o0;  kb[base + 1] = (bf16)o1;
            }
        }
    }
    for (int e = threadIdx.x; e < kNKV * kHD; e += blockDim.x) {
        int kh = e / kHD, d = e % kHD;
        float v = row[(kNH + kNKV) * kHD + e];
        for (int r = 0; r < kG; ++r) {
            int h = kh * kG + r;
            vt[(((long)b * kNH + h) * kHD + d) * kS + spos] = (bf16)v;
        }
    }
}

__global__ __launch_bounds__(256)
void softmax_kernel(const float* __restrict__ scores, bf16* __restrict__ probs,
                    float scale)
{
    long rowi = blockIdx.x;
    int r = (int)(rowi % kS);
    const float* srow = scores + rowi * (long)kS;
    bf16* prow = probs + rowi * (long)kS;
    int valid = r + 1;
    int lane = threadIdx.x & 31, wave = threadIdx.x >> 5;
    __shared__ float sh[9];

    float mx = -3.0e38f;
    for (int i = threadIdx.x; i < valid; i += 256) mx = fmaxf(mx, srow[i] * scale);
    for (int o = 16; o > 0; o >>= 1) mx = fmaxf(mx, __shfl_down(mx, o, 32));
    if (lane == 0) sh[wave] = mx;
    __syncthreads();
    if (wave == 0) {
        float v = (lane < 8) ? sh[lane] : -3.0e38f;
        for (int o = 4; o > 0; o >>= 1) v = fmaxf(v, __shfl_down(v, o, 32));
        if (lane == 0) sh[8] = v;
    }
    __syncthreads();
    float m = sh[8];

    float sum = 0.f;
    for (int i = threadIdx.x; i < valid; i += 256) sum += __expf(srow[i] * scale - m);
    for (int o = 16; o > 0; o >>= 1) sum += __shfl_down(sum, o, 32);
    if (lane == 0) sh[wave] = sum;
    __syncthreads();
    if (wave == 0) {
        float v = (lane < 8) ? sh[lane] : 0.f;
        for (int o = 4; o > 0; o >>= 1) v += __shfl_down(v, o, 32);
        if (lane == 0) sh[8] = v;
    }
    __syncthreads();
    float inv = 1.0f / sh[8];

    for (int i = threadIdx.x; i < kS; i += 256) {
        float p = (i < valid) ? __expf(srow[i] * scale - m) * inv : 0.0f;
        prow[i] = (bf16)p;
    }
}

__global__ __launch_bounds__(256)
void merge_heads_kernel(const float* __restrict__ ao, bf16* __restrict__ ob)
{
    int n = blockIdx.x;
    int b = n / kS, spos = n % kS;
    for (int i = threadIdx.x; i < kD; i += blockDim.x) {
        int h = i / kHD, d = i % kHD;
        ob[(long)n * kD + i] =
            (bf16)ao[(((long)b * kNH + h) * kS + spos) * kHD + d];
    }
}

__global__ void silu_mul_kernel(const float* __restrict__ g1,
                                const float* __restrict__ g2,
                                bf16* __restrict__ out, long n)
{
    long i = (long)blockIdx.x * blockDim.x + threadIdx.x;
    if (i < n) {
        float a = g1[i];
        float s = a / (1.0f + __expf(-a));
        out[i] = (bf16)(s * g2[i]);
    }
}

// ---------------------------------------------------------------------------
// Launcher
// ---------------------------------------------------------------------------
extern "C" void kernel_launch(void* const* d_in, const int* in_sizes, int n_in,
                              void* d_out, int out_size, void* d_ws, size_t ws_size,
                              hipStream_t stream)
{
    (void)in_sizes; (void)n_in; (void)out_size; (void)ws_size;
    const int*   ids   = (const int*)  d_in[0];
    // d_in[1] = input_pos (arange(S)) -> implied by token index
    const float* emb   = (const float*)d_in[2];
    const float* w_qkv = (const float*)d_in[3];
    const float* w_fc  = (const float*)d_in[4];
    const float* w1    = (const float*)d_in[5];
    const float* w2    = (const float*)d_in[6];
    const float* w3    = (const float*)d_in[7];
    const float* anw   = (const float*)d_in[8];
    const float* fnw   = (const float*)d_in[9];
    const float* nw    = (const float*)d_in[10];
    const float* lm    = (const float*)d_in[11];
    float* out = (float*)d_out;

    char* ws = (char*)d_ws;
    size_t off = 0;
    auto take = [&](size_t bytes) -> void* {
        void* p = ws + off;
        off += (bytes + 255) & ~(size_t)255;
        return p;
    };

    bf16*  wqkv_b = (bf16*) take((size_t)kL * kQKV * kD * 2);
    bf16*  wfc_b  = (bf16*) take((size_t)kL * kD * kD * 2);
    bf16*  w1_b   = (bf16*) take((size_t)kL * kF * kD * 2);
    bf16*  w2_b   = (bf16*) take((size_t)kL * kF * kD * 2);
    bf16*  w3_b   = (bf16*) take((size_t)kL * kD * kF * 2);
    bf16*  lm_b   = (bf16*) take((size_t)kV * kD * 2);
    float* x      = (float*)take((size_t)kNT * kD * 4);
    bf16*  hb     = (bf16*) take((size_t)kNT * kD * 2);
    float* qkv    = (float*)take((size_t)kNT * kQKV * 4);
    bf16*  qb     = (bf16*) take((size_t)kB * kNH * kS * kHD * 2);
    bf16*  kb     = (bf16*) take((size_t)kB * kNH * kS * kHD * 2);
    bf16*  vt     = (bf16*) take((size_t)kB * kNH * kS * kHD * 2);
    bf16*  pb     = (bf16*) take((size_t)kB * kNH * kS * kS * 2);
    float* ao     = (float*)take((size_t)kB * kNH * kS * kHD * 4);
    bf16*  ob     = (bf16*) take((size_t)kNT * kD * 2);
    float* scores = (float*)take((size_t)kB * kNH * kS * kS * 4);
    // FFN scratch aliases the (larger, then-dead) scores buffer
    float* g1 = scores;
    float* g2 = scores + (size_t)kNT * kF;
    bf16*  gl = (bf16*)(g2 + (size_t)kNT * kF);

    auto cvt = [&](const float* src, bf16* dst, size_t n) {
        unsigned blocks = (unsigned)((n / 4 + 255) / 256);
        cvt_f32_bf16_kernel<<<blocks, 256, 0, stream>>>(src, dst, (long)n);
    };
    auto gemm = [&](float* C, const bf16* A, const bf16* W,
                    int M, int N, int K, int batch,
                    long sC, long sA, long sW2, int accum, int causal) {
        dim3 grid((N + BN - 1) / BN, (M + BM - 1) / BM, batch);
        gemm_bf16_kernel<<<grid, dim3(256), 0, stream>>>(
            C, A, W, M, N, K, sC, sA, sW2, accum, causal);
    };

    // Weight conversion fp32 -> bf16 (streamed once per call)
    cvt(w_qkv, wqkv_b, (size_t)kL * kQKV * kD);
    cvt(w_fc,  wfc_b,  (size_t)kL * kD * kD);
    cvt(w1,    w1_b,   (size_t)kL * kF * kD);
    cvt(w2,    w2_b,   (size_t)kL * kF * kD);
    cvt(w3,    w3_b,   (size_t)kL * kD * kF);
    cvt(lm,    lm_b,   (size_t)kV * kD);

    embed_kernel<<<kNT, 256, 0, stream>>>(ids, emb, x);

    const float scale = 1.0f / sqrtf((float)kD);   // reference scales by 1/sqrt(D)

    for (int l = 0; l < kL; ++l) {
        rmsnorm_kernel<<<kNT, 256, 0, stream>>>(x, anw + (long)l * kD, hb);
        gemm(qkv, hb, wqkv_b + (size_t)l * kQKV * kD,
             kNT, kQKV, kD, 1, 0, 0, 0, 0, 0);
        rope_split_kernel<<<kNT, 256, 0, stream>>>(qkv, qb, kb, vt);
        // scores[b,h] = q @ k^T  (causal: skip tiles above diagonal)
        gemm(scores, qb, kb, kS, kS, kHD, kB * kNH,
             (long)kS * kS, (long)kS * kHD, (long)kS * kHD, 0, 1);
        softmax_kernel<<<kB * kNH * kS, 256, 0, stream>>>(scores, pb, scale);
        // o[b,h] = attn @ v
        gemm(ao, pb, vt, kS, kHD, kS, kB * kNH,
             (long)kS * kHD, (long)kS * kS, (long)kHD * kS, 0, 0);
        merge_heads_kernel<<<kNT, 256, 0, stream>>>(ao, ob);
        // x += o @ w_fc^T
        gemm(x, ob, wfc_b + (size_t)l * kD * kD, kNT, kD, kD, 1, 0, 0, 0, 1, 0);

        rmsnorm_kernel<<<kNT, 256, 0, stream>>>(x, fnw + (long)l * kD, hb);
        gemm(g1, hb, w1_b + (size_t)l * kF * kD, kNT, kF, kD, 1, 0, 0, 0, 0, 0);
        gemm(g2, hb, w2_b + (size_t)l * kF * kD, kNT, kF, kD, 1, 0, 0, 0, 0, 0);
        long ntot = (long)kNT * kF;
        silu_mul_kernel<<<(unsigned)((ntot + 255) / 256), 256, 0, stream>>>(
            g1, g2, gl, ntot);
        // x += (silu(g1)*g2) @ w3^T
        gemm(x, gl, w3_b + (size_t)l * kD * kF, kNT, kD, kF, 1, 0, 0, 0, 1, 0);
    }

    rmsnorm_kernel<<<kNT, 256, 0, stream>>>(x, nw, hb);
    // logits = h @ lm_head^T
    gemm(out, hb, lm_b, kNT, kV, kD, 1, 0, 0, 0, 0, 0);
}
